// USSPA_G_T1_65326452572274
// MI455X (gfx1250) — matrix operand order repaired and checked by reading, not compiled, not run on hardware
//
#include <hip/hip_runtime.h>
#include <hip/hip_bf16.h>
#include <cstdint>
#include <cstddef>

// ---------------------------------------------------------------------------
// USSPA-style point completion for MI455X (gfx1250, wave32, WMMA).
// All matmuls use v_wmma_f32_16x16x32_f16 (f16 in, f32 accumulate).
// GEMM: each wave owns a (MT x 64) output tile (MT=32 for the big per-point
//       layers -> B fragments reused across two A row-strips, ~22 flop/B from
//       L2; MT=16 for the 16-row decoder layers).
// FPS: 1 workgroup per batch, register-resident per-thread distances,
//      wave32 __shfl_down argmax reduction (3 barriers / iteration).
// Workspace use: ~385 MB (h1/h2/h3/h4 activation tensors + f16 weights).
// ---------------------------------------------------------------------------

typedef __attribute__((ext_vector_type(16))) _Float16 v16h;
typedef __attribute__((ext_vector_type(8)))  _Float16 v8h;
typedef __attribute__((ext_vector_type(8)))  float    v8f;

#define NBATCH 16            // 2 * 8 (partial ++ complete)
#define NPT    8192          // points per batch
#define GPN_C  512
#define NCAND  (2*NPT + GPN_C)   // 16896 candidate points per batch
#define KOUT   2048
#define MROWS  (NBATCH*NPT)  // 131072 rows for per-point MLPs

// ---------------- small utility kernels ----------------

__global__ void f32_to_f16_kernel(const float* __restrict__ in,
                                  _Float16* __restrict__ out, int n) {
  int i = blockIdx.x * blockDim.x + threadIdx.x;
  if (i < n) out[i] = (_Float16)in[i];
}

__global__ void concat_x_kernel(const float* __restrict__ a,
                                const float* __restrict__ b,
                                float* __restrict__ x) {
  int i = blockIdx.x * blockDim.x + threadIdx.x;
  const int half = 8 * NPT * 3;
  if (i < half) x[i] = a[i];
  else if (i < 2 * half) x[i] = b[i - half];
}

// e1: K=3 is too small for WMMA; pure VALU, f32 math, f16 out. relu.
__global__ void e1_kernel(const float* __restrict__ x,
                          const float* __restrict__ w,   // (128,3) f32
                          const float* __restrict__ bias,
                          _Float16* __restrict__ out) {  // (MROWS,128)
  int i = blockIdx.x * blockDim.x + threadIdx.x;
  if (i >= MROWS * 128) return;
  int m = i >> 7, c = i & 127;
  float v = bias[c] + x[m*3+0]*w[c*3+0] + x[m*3+1]*w[c*3+1] + x[m*3+2]*w[c*3+2];
  out[i] = (_Float16)(v > 0.f ? v : 0.f);
}

// ---------------- generic WMMA GEMM ----------------
// O[M,N] = act( A[M,K] @ W[N,K]^T + bias[N] + rowbias[row>>rbShift][N] )
// A,W f16 row-major. Each wave computes an (MT x 64) strip.
// Requires M%MT==0, N%64==0, K%32==0.
template<int MT>   // 16 or 32
__global__ __launch_bounds__(256)
void gemm_f16_wmma(const _Float16* __restrict__ A, int lda,
                   const _Float16* __restrict__ W, int ldw,
                   const float* __restrict__ bias,     // nullable
                   const float* __restrict__ rowbias,  // nullable, [M>>rbShift][N]
                   int rbShift,
                   _Float16* __restrict__ O16,         // required
                   float* __restrict__ O32,            // nullable
                   int ldo, int M, int N, int K, int relu)
{
  constexpr int MI = MT / 16;               // A row-strips per wave
  const int lane = threadIdx.x & 31;
  const int wave = threadIdx.x >> 5;
  const int tilesN = N >> 6;
  const int tilesM = M / MT;
  const int waveId = blockIdx.x * 8 + wave;
  const int tM = waveId / tilesN;
  const int tN = waveId - tM * tilesN;
  if (tM >= tilesM) return;                 // whole-wave uniform exit
  const int row0 = tM * MT;
  const int col0 = tN << 6;
  const int hi = lane >> 4;                 // half-wave select
  const int lh = lane & 15;

  v8f acc[MI][4];
  #pragma unroll
  for (int im = 0; im < MI; ++im)
    #pragma unroll
    for (int t = 0; t < 4; ++t)
      #pragma unroll
      for (int r = 0; r < 8; ++r) acc[im][t][r] = 0.f;

  for (int k0 = 0; k0 < K; k0 += 32) {
    // --- A fragments: 16-bit A 16x32 layout ---
    // lane(0..15): row M=lane, K = hi*16 + [0..15] (contiguous)
    union AF { v16h v; v8h h[2]; } af[MI];
    #pragma unroll
    for (int im = 0; im < MI; ++im) {
      const _Float16* ap = A + (size_t)(row0 + im * 16 + lh) * lda + k0;
      af[im].h[0] = *(const v8h*)(ap + hi * 16);
      af[im].h[1] = *(const v8h*)(ap + hi * 16 + 8);
    }

    #pragma unroll
    for (int t = 0; t < 4; ++t) {
      // --- B fragment: 16-bit B 32x16 layout ---
      // lane: col N = col0 + t*16 + lh, K = k0 + hi*16 + [0..15] (contiguous)
      const _Float16* wp = W + (size_t)(col0 + t * 16 + lh) * ldw + k0 + hi * 16;
      v16h bf = *(const v16h*)(wp);
      #pragma unroll
      for (int im = 0; im < MI; ++im)
        acc[im][t] = __builtin_amdgcn_wmma_f32_16x16x32_f16(
                       false, af[im].v, false, bf, (short)0, acc[im][t], false, false);
    }
  }

  // epilogue: C/D layout — VGPR r: row = hi*8 + r, col = lane&15
  #pragma unroll
  for (int im = 0; im < MI; ++im)
    #pragma unroll
    for (int t = 0; t < 4; ++t)
      #pragma unroll
      for (int r = 0; r < 8; ++r) {
        int row = row0 + im * 16 + hi * 8 + r;
        int col = col0 + t * 16 + lh;
        float v = acc[im][t][r];
        if (bias)    v += bias[col];
        if (rowbias) v += rowbias[(row >> rbShift) * N + col];
        if (relu)    v = v > 0.f ? v : 0.f;
        size_t o = (size_t)row * ldo + col;
        O16[o] = (_Float16)v;
        if (O32) O32[o] = v;
      }
}

// ---------------- per-batch column-wise max over points ----------------
// H is (NBATCH, rows, C) f16 contiguous. blockIdx.x = batch, threadIdx.x = channel.
__global__ void rowmax_kernel(const _Float16* __restrict__ H, int rows, int C,
                              float* __restrict__ of32, _Float16* __restrict__ of16) {
  int b = blockIdx.x, c = threadIdx.x;
  const _Float16* p = H + (size_t)b * rows * C + c;
  float m = -3.0e38f;
  #pragma unroll 8
  for (int r = 0; r < rows; ++r) m = fmaxf(m, (float)p[(size_t)r * C]);
  if (of32) of32[b * C + c] = m;
  if (of16) of16[b * C + c] = (_Float16)m;
}

// ---------------- mirror head: ab = tanh(m2a @ m3_w^T + m3_b) ----------------
__global__ void mirror_head_kernel(const _Float16* __restrict__ m2a, // (16,128)
                                   const float* __restrict__ w,      // (2,128) f32
                                   const float* __restrict__ bias,   // (2)
                                   float* __restrict__ AB) {         // (16,2)
  int t = threadIdx.x;
  if (t < 32) {
    int b = t >> 1, o = t & 1;
    float s = bias[o];
    for (int k = 0; k < 128; ++k) s += (float)m2a[b * 128 + k] * w[o * 128 + k];
    AB[b * 2 + o] = tanhf(s);
  }
}

// ---------------- build candidate set: [x ; mirrored(x) ; point_R_0] ----------------
__global__ void build_candidates_kernel(const float* __restrict__ x,   // (16,8192,3)
                                        const float* __restrict__ AB,  // (16,2)
                                        const float* __restrict__ pr0, // (16,1536)
                                        float* __restrict__ cand) {    // (16,16896,3)
  int i = blockIdx.x * blockDim.x + threadIdx.x;
  if (i >= NBATCH * NCAND) return;
  int b = i / NCAND;
  int p = i - b * NCAND;
  float ox, oy, oz;
  if (p < NPT) {
    const float* s = x + ((size_t)b * NPT + p) * 3;
    ox = s[0]; oy = s[1]; oz = s[2];
  } else if (p < 2 * NPT) {
    const float* s = x + ((size_t)b * NPT + (p - NPT)) * 3;
    float px = s[0], py = s[1], pz = s[2];
    float A = AB[b * 2 + 0], Bc = AB[b * 2 + 1];
    float t = 2.0f * (A * px + Bc * pz) / (A * A + Bc * Bc + 1e-8f);
    ox = px - A * t; oy = py; oz = pz - Bc * t;
  } else {
    int q = p - 2 * NPT;
    const float* s = pr0 + (size_t)b * (GPN_C * 3) + q * 3;
    ox = s[0]; oy = s[1]; oz = s[2];
  }
  float* d = cand + (size_t)i * 3;
  d[0] = ox; d[1] = oy; d[2] = oz;
}

// ---------------- farthest point sampling ----------------
// One workgroup (1024 threads) per batch. Distances live in registers
// (17 per thread); argmax via wave32 shuffles + 32-entry LDS stage.
__global__ __launch_bounds__(1024)
void fps_kernel(const float* __restrict__ cand, float* __restrict__ out) {
  const int b = blockIdx.x;
  const float* P = cand + (size_t)b * NCAND * 3;
  float* O = out + (size_t)b * KOUT * 3;
  const int tid = threadIdx.x;
  const int PPT = 17;                     // ceil(16896/1024)
  float dist[PPT];
  #pragma unroll
  for (int j = 0; j < PPT; ++j) dist[j] = 1.0e10f;

  __shared__ float s_val[32];
  __shared__ int   s_idx[32];
  __shared__ int   s_last;
  __shared__ float s_p[3];

  if (tid < 3) O[tid] = P[tid];           // out[0] = points[0]
  if (tid == 0) s_last = 0;
  __syncthreads();
  int last = 0;

  for (int it = 1; it < KOUT; ++it) {
    if (tid < 3) s_p[tid] = P[(size_t)last * 3 + tid];
    __syncthreads();
    float px = s_p[0], py = s_p[1], pz = s_p[2];

    float bestV = -1.0f; int bestI = NCAND;
    #pragma unroll
    for (int j = 0; j < PPT; ++j) {
      int i = tid + (j << 10);
      if (i < NCAND) {
        const float* q = P + (size_t)i * 3;
        float dx = q[0] - px, dy = q[1] - py, dz = q[2] - pz;
        float d = dx * dx + dy * dy + dz * dz;
        float nd = fminf(dist[j], d);
        dist[j] = nd;
        if (nd > bestV || (nd == bestV && i < bestI)) { bestV = nd; bestI = i; }
      }
    }
    // wave32 argmax (first-index tie-break like jnp.argmax)
    #pragma unroll
    for (int off = 16; off > 0; off >>= 1) {
      float ov = __shfl_down(bestV, off, 32);
      int   oi = __shfl_down(bestI, off, 32);
      if (ov > bestV || (ov == bestV && oi < bestI)) { bestV = ov; bestI = oi; }
    }
    if ((tid & 31) == 0) { s_val[tid >> 5] = bestV; s_idx[tid >> 5] = bestI; }
    __syncthreads();
    if (tid < 32) {
      bestV = s_val[tid]; bestI = s_idx[tid];
      #pragma unroll
      for (int off = 16; off > 0; off >>= 1) {
        float ov = __shfl_down(bestV, off, 32);
        int   oi = __shfl_down(bestI, off, 32);
        if (ov > bestV || (ov == bestV && oi < bestI)) { bestV = ov; bestI = oi; }
      }
      if (tid == 0) {
        s_last = bestI;
        const float* q = P + (size_t)bestI * 3;
        O[(size_t)it * 3 + 0] = q[0];
        O[(size_t)it * 3 + 1] = q[1];
        O[(size_t)it * 3 + 2] = q[2];
      }
    }
    __syncthreads();
    last = s_last;
  }
}

// ---------------------------------------------------------------------------
// host launcher
// ---------------------------------------------------------------------------
extern "C" void kernel_launch(void* const* d_in, const int* in_sizes, int n_in,
                              void* d_out, int out_size, void* d_ws, size_t ws_size,
                              hipStream_t stream) {
  const float* partial = (const float*)d_in[0];
  const float* complete = (const float*)d_in[1];
  const float* e1_w = (const float*)d_in[2];  const float* e1_b = (const float*)d_in[3];
  const float* e2_w = (const float*)d_in[4];  const float* e2_b = (const float*)d_in[5];
  const float* e3_w = (const float*)d_in[6];  const float* e3_b = (const float*)d_in[7];
  const float* e4_w = (const float*)d_in[8];  const float* e4_b = (const float*)d_in[9];
  const float* d1_w = (const float*)d_in[10]; const float* d1_b = (const float*)d_in[11];
  const float* d2_w = (const float*)d_in[12]; const float* d2_b = (const float*)d_in[13];
  const float* d3_w = (const float*)d_in[14]; const float* d3_b = (const float*)d_in[15];
  const float* d4_w = (const float*)d_in[16]; const float* d4_b = (const float*)d_in[17];
  const float* d5_w = (const float*)d_in[18]; const float* d5_b = (const float*)d_in[19];
  const float* m1_w = (const float*)d_in[20]; const float* m1_b = (const float*)d_in[21];
  const float* m2_w = (const float*)d_in[22]; const float* m2_b = (const float*)d_in[23];
  const float* m3_w = (const float*)d_in[24]; const float* m3_b = (const float*)d_in[25];
  float* out = (float*)d_out;

  char* ws = (char*)d_ws;
  size_t off = 0;
  auto alloc = [&](size_t bytes) -> char* {
    off = (off + 255) & ~(size_t)255;
    char* p = ws + off;
    off += bytes;
    return p;
  };

  // f32 workspace
  float* x_f32 = (float*)alloc((size_t)MROWS * 3 * 4);
  float* gbias = (float*)alloc((size_t)NBATCH * 512 * 4);
  float* f_f32 = (float*)alloc((size_t)NBATCH * 512 * 4);
  float* pr0   = (float*)alloc((size_t)NBATCH * 1536 * 4);
  float* AB    = (float*)alloc((size_t)NBATCH * 2 * 4);
  float* cand  = (float*)alloc((size_t)NBATCH * NCAND * 3 * 4);
  // f16 weights
  _Float16* w2  = (_Float16*)alloc((size_t)256 * 128 * 2);
  _Float16* w3  = (_Float16*)alloc((size_t)512 * 512 * 2);
  _Float16* w4  = (_Float16*)alloc((size_t)512 * 512 * 2);
  _Float16* wd1 = (_Float16*)alloc((size_t)512 * 512 * 2);
  _Float16* wd2 = (_Float16*)alloc((size_t)512 * 512 * 2);
  _Float16* wd3 = (_Float16*)alloc((size_t)1024 * 512 * 2);
  _Float16* wd4 = (_Float16*)alloc((size_t)1024 * 1024 * 2);
  _Float16* wd5 = (_Float16*)alloc((size_t)1536 * 1024 * 2);
  _Float16* wm1 = (_Float16*)alloc((size_t)128 * 512 * 2);
  _Float16* wm2 = (_Float16*)alloc((size_t)128 * 128 * 2);
  // f16 activations
  _Float16* g_f16 = (_Float16*)alloc((size_t)NBATCH * 256 * 2);
  _Float16* f_f16 = (_Float16*)alloc((size_t)NBATCH * 512 * 2);
  _Float16* sink  = (_Float16*)alloc((size_t)NBATCH * 1536 * 2); // f16 sink for gbias
  _Float16* a1    = (_Float16*)alloc((size_t)NBATCH * 512 * 2);
  _Float16* a2    = (_Float16*)alloc((size_t)NBATCH * 512 * 2);
  _Float16* a3    = (_Float16*)alloc((size_t)NBATCH * 1024 * 2);
  _Float16* a4    = (_Float16*)alloc((size_t)NBATCH * 1024 * 2);
  _Float16* a5    = (_Float16*)alloc((size_t)NBATCH * 1536 * 2);
  _Float16* am1   = (_Float16*)alloc((size_t)NBATCH * 128 * 2);
  _Float16* am2   = (_Float16*)alloc((size_t)NBATCH * 128 * 2);
  _Float16* h1 = (_Float16*)alloc((size_t)MROWS * 128 * 2);   // 33.5 MB
  _Float16* h2 = (_Float16*)alloc((size_t)MROWS * 256 * 2);   // 67 MB
  _Float16* h3 = (_Float16*)alloc((size_t)MROWS * 512 * 2);   // 134 MB
  _Float16* h4 = (_Float16*)alloc((size_t)MROWS * 512 * 2);   // 134 MB

  auto cvt = [&](const float* src, _Float16* dst, int n) {
    f32_to_f16_kernel<<<(n + 255) / 256, 256, 0, stream>>>(src, dst, n);
  };
  cvt(e2_w, w2, 256 * 128);
  cvt(e3_w, w3, 512 * 512);
  cvt(e4_w, w4, 512 * 512);
  cvt(d1_w, wd1, 512 * 512);
  cvt(d2_w, wd2, 512 * 512);
  cvt(d3_w, wd3, 1024 * 512);
  cvt(d4_w, wd4, 1024 * 1024);
  cvt(d5_w, wd5, 1536 * 1024);
  cvt(m1_w, wm1, 128 * 512);
  cvt(m2_w, wm2, 128 * 128);

  concat_x_kernel<<<(MROWS * 3 + 255) / 256, 256, 0, stream>>>(partial, complete, x_f32);

  auto gemm = [&](const _Float16* A, int lda, const _Float16* W, int ldw,
                  const float* bias, const float* rowbias, int rbShift,
                  _Float16* o16, float* o32, int M, int N, int K, int relu) {
    if ((M % 32) == 0) {
      int tiles = (M / 32) * (N / 64);
      gemm_f16_wmma<32><<<(tiles + 7) / 8, 256, 0, stream>>>(
          A, lda, W, ldw, bias, rowbias, rbShift, o16, o32, N, M, N, K, relu);
    } else {
      int tiles = (M / 16) * (N / 64);
      gemm_f16_wmma<16><<<(tiles + 7) / 8, 256, 0, stream>>>(
          A, lda, W, ldw, bias, rowbias, rbShift, o16, o32, N, M, N, K, relu);
    }
  };

  // encoder
  e1_kernel<<<(MROWS * 128 + 255) / 256, 256, 0, stream>>>(x_f32, e1_b ? e1_w : e1_w, e1_b, h1);
  gemm(h1, 128, w2, 128, e2_b, nullptr, 0, h2, nullptr, MROWS, 256, 128, 0);
  rowmax_kernel<<<NBATCH, 256, 0, stream>>>(h2, NPT, 256, nullptr, g_f16);
  // per-batch bias for e3: gbias = g @ e3_w[:,256:]^T + e3_b
  gemm(g_f16, 256, w3 + 256, 512, e3_b, nullptr, 0, sink, gbias, NBATCH, 512, 256, 0);
  // e3 main: h3 = relu(h2 @ e3_w[:,:256]^T + gbias[batch])
  gemm(h2, 256, w3, 512, nullptr, gbias, 13, h3, nullptr, MROWS, 512, 256, 1);
  gemm(h3, 512, w4, 512, e4_b, nullptr, 0, h4, nullptr, MROWS, 512, 512, 0);
  rowmax_kernel<<<NBATCH, 512, 0, stream>>>(h4, NPT, 512, f_f32, f_f16);

  // decoder (M = 16 == one WMMA tile)
  gemm(f_f16, 512, wd1, 512, d1_b, nullptr, 0, a1, nullptr, NBATCH, 512, 512, 1);
  gemm(a1, 512, wd2, 512, d2_b, nullptr, 0, a2, nullptr, NBATCH, 512, 512, 1);
  gemm(a2, 512, wd3, 512, d3_b, nullptr, 0, a3, nullptr, NBATCH, 1024, 512, 1);
  gemm(a3, 1024, wd4, 1024, d4_b, nullptr, 0, a4, nullptr, NBATCH, 1024, 1024, 1);
  gemm(a4, 1024, wd5, 1024, d5_b, nullptr, 0, a5, pr0, NBATCH, 1536, 1024, 0);

  // mirror head
  gemm(f_f16, 512, wm1, 512, m1_b, nullptr, 0, am1, nullptr, NBATCH, 128, 512, 1);
  gemm(am1, 128, wm2, 128, m2_b, nullptr, 0, am2, nullptr, NBATCH, 128, 128, 1);
  mirror_head_kernel<<<1, 32, 0, stream>>>(am2, m3_w, m3_b, AB);

  // candidates + FPS
  build_candidates_kernel<<<(NBATCH * NCAND + 255) / 256, 256, 0, stream>>>(
      x_f32, AB, pr0, cand);
  fps_kernel<<<NBATCH, 1024, 0, stream>>>(cand, out);
}
